// HLLMTransformerBlock_62697932587244
// MI455X (gfx1250) — matrix operand order repaired and checked by reading, compiled, not run on hardware
//
#include <hip/hip_runtime.h>
#include <hip/hip_bf16.h>

typedef __bf16 bf16_t;
typedef __attribute__((ext_vector_type(16))) __bf16 v16bf;
typedef __attribute__((ext_vector_type(8)))  __bf16 v8bf;
typedef __attribute__((ext_vector_type(8)))  float  v8f;

#define TB_B   4
#define TB_L   2048
#define TB_D   1024
#define TB_H   16
#define TB_HD  64
#define TB_FFN 4096
#define TB_M   (TB_B * TB_L)

__device__ __forceinline__ bf16_t f2bf(float x) {
    union { float f; unsigned u; } c; c.f = x;
    unsigned r = c.u + 0x7FFFu + ((c.u >> 16) & 1u);
    unsigned short hs = (unsigned short)(r >> 16);
    bf16_t b;
    __builtin_memcpy(&b, &hs, 2);
    return b;
}

__device__ __forceinline__ v8f wmma_bf16(v16bf a, v16bf b, v8f c) {
    return __builtin_amdgcn_wmma_f32_16x16x32_bf16(false, a, false, b, (short)0, c, false, false);
}

// Async global -> LDS copy of 16 bytes (ASYNCcnt-tracked, no VGPR staging).
__device__ __forceinline__ void async_copy_b128(unsigned ldsAddr, const void* gptr) {
    asm volatile("global_load_async_to_lds_b128 %0, %1, off"
                 :: "v"(ldsAddr), "v"((unsigned long long)(size_t)gptr)
                 : "memory");
}

// ---------------------------------------------------------------------------
// Weight convert: w [K,N] fp32  ->  wT [N,K] bf16
// ---------------------------------------------------------------------------
__global__ void tb_convert_wT(const float* __restrict__ w, bf16_t* __restrict__ wT,
                              int K, int N) {
    long long idx = (long long)blockIdx.x * blockDim.x + threadIdx.x;
    long long tot = (long long)K * N;
    if (idx >= tot) return;
    int n = (int)(idx / K);
    int k = (int)(idx % K);
    wT[idx] = f2bf(w[(long long)k * N + n]);
}

// ---------------------------------------------------------------------------
// LayerNorm over last dim (D=1024), fp32 in -> bf16 out. One block per row.
// ---------------------------------------------------------------------------
__global__ __launch_bounds__(256) void tb_layernorm(const float* __restrict__ x,
                                                    const float* __restrict__ g,
                                                    const float* __restrict__ b,
                                                    bf16_t* __restrict__ xn) {
    __shared__ float red[256];
    int tid = threadIdx.x;
    long long row = blockIdx.x;
    const float* xr = x + row * TB_D;

    float vals[4];
    float s = 0.f;
#pragma unroll
    for (int i = 0; i < 4; ++i) { vals[i] = xr[tid + i * 256]; s += vals[i]; }
    red[tid] = s; __syncthreads();
    for (int o = 128; o > 0; o >>= 1) { if (tid < o) red[tid] += red[tid + o]; __syncthreads(); }
    float mu = red[0] * (1.0f / TB_D);
    __syncthreads();

    s = 0.f;
#pragma unroll
    for (int i = 0; i < 4; ++i) { float d = vals[i] - mu; s += d * d; }
    red[tid] = s; __syncthreads();
    for (int o = 128; o > 0; o >>= 1) { if (tid < o) red[tid] += red[tid + o]; __syncthreads(); }
    float rstd = rsqrtf(red[0] * (1.0f / TB_D) + 1e-5f);
    __syncthreads();

#pragma unroll
    for (int i = 0; i < 4; ++i) {
        int col = tid + i * 256;
        xn[row * TB_D + col] = f2bf((vals[i] - mu) * rstd * g[col] + b[col]);
    }
}

// ---------------------------------------------------------------------------
// v [B,L,D] bf16 -> vT [B,H,HD,L] bf16  (indexed over output, coalesced write)
// ---------------------------------------------------------------------------
__global__ void tb_transpose_v(const bf16_t* __restrict__ v, bf16_t* __restrict__ vT) {
    long long o = (long long)blockIdx.x * blockDim.x + threadIdx.x;
    long long tot = (long long)TB_B * TB_D * TB_L;
    if (o >= tot) return;
    int l   = (int)(o % TB_L);
    long long r = o / TB_L;
    int d   = (int)(r % TB_HD);
    long long bh = r / TB_HD;
    int h   = (int)(bh % TB_H);
    int bb  = (int)(bh / TB_H);
    vT[o] = v[((long long)bb * TB_L + l) * TB_D + h * TB_HD + d];
}

// ---------------------------------------------------------------------------
// GEMM: C[M,N] = A[M,K](bf16) * W[K,N] (stored transposed BT[N,K] bf16)
//       + bias, optional residual(fp32), optional ReLU; out fp32 or bf16.
// Block tile 128x128, BK=32, 256 threads = 8 waves, wave tile 32x64.
// Double-buffered LDS fed by GLOBAL_LOAD_ASYNC_TO_LDS_B128 (ASYNCcnt).
// ---------------------------------------------------------------------------
template <bool OUT_F32, bool RELU, bool ADD_RES>
__global__ __launch_bounds__(256) void tb_gemm(const bf16_t* __restrict__ A,
                                               const bf16_t* __restrict__ BT,
                                               const float* __restrict__ bias,
                                               const float* __restrict__ res,
                                               float* __restrict__ outF,
                                               bf16_t* __restrict__ outB,
                                               int M, int N, int K) {
    __shared__ bf16_t Al[2][128][32];
    __shared__ bf16_t Bl[2][128][32];

    const int tid  = threadIdx.x;
    const int lane = tid & 31;
    const int wave = tid >> 5;
    const int half = lane >> 4;
    const int lr   = lane & 15;

    const int m0 = blockIdx.y * 128;
    const int n0 = blockIdx.x * 128;
    const int wm = (wave & 3) * 32;
    const int wn = (wave >> 2) * 64;

    v8f acc[2][4];
#pragma unroll
    for (int i = 0; i < 2; ++i)
#pragma unroll
        for (int j = 0; j < 4; ++j)
            acc[i][j] = (v8f){0.f, 0.f, 0.f, 0.f, 0.f, 0.f, 0.f, 0.f};

    const int arow = tid >> 1;          // 0..127, each thread copies 32B of A and B
    const int ak   = (tid & 1) * 16;    // element offset (16 bf16 = 32B)

    auto issueTile = [&](int k0, int buf) {
        const bf16_t* ga = A + (long long)(m0 + arow) * K + k0 + ak;
        unsigned la = (unsigned)(size_t)&Al[buf][arow][ak];
        async_copy_b128(la,      ga);
        async_copy_b128(la + 16, ga + 8);
        const bf16_t* gb = BT + (long long)(n0 + arow) * K + k0 + ak;
        unsigned lb = (unsigned)(size_t)&Bl[buf][arow][ak];
        async_copy_b128(lb,      gb);
        async_copy_b128(lb + 16, gb + 8);
    };

    issueTile(0, 0);

    int buf = 0;
    for (int k0 = 0; k0 < K; k0 += 32, buf ^= 1) {
        if (k0 + 32 < K) {
            issueTile(k0 + 32, buf ^ 1);
            // 8 async ops outstanding; older 4 (current tile) complete in order.
            asm volatile("s_wait_asynccnt 0x4" ::: "memory");
        } else {
            asm volatile("s_wait_asynccnt 0x0" ::: "memory");
        }
        __syncthreads();   // publish all waves' LDS tile writes

        v16bf af[2], bfr[4];
#pragma unroll
        for (int i = 0; i < 2; ++i) {
            int row = wm + i * 16 + lr;
            union { v16bf v; v8bf h[2]; } u;
            u.h[0] = *(const v8bf*)(&Al[buf][row][half * 16]);
            u.h[1] = *(const v8bf*)(&Al[buf][row][half * 16 + 8]);
            af[i] = u.v;
        }
#pragma unroll
        for (int j = 0; j < 4; ++j) {
            int col = wn + j * 16 + lr;
            bfr[j] = *(const v16bf*)(&Bl[buf][col][half * 16]);
        }
#pragma unroll
        for (int i = 0; i < 2; ++i)
#pragma unroll
            for (int j = 0; j < 4; ++j)
                acc[i][j] = wmma_bf16(af[i], bfr[j], acc[i][j]);
        // Note: next iteration's issueTile targets the buffer read in
        // this iteration; no barrier separates these reads from that
        // write, so a slower wave may still be reading it.
    }

#pragma unroll
    for (int i = 0; i < 2; ++i) {
#pragma unroll
        for (int j = 0; j < 4; ++j) {
            int col = n0 + wn + j * 16 + lr;
            float bc = bias[col];
#pragma unroll
            for (int v = 0; v < 8; ++v) {
                int row = m0 + wm + i * 16 + v + half * 8;
                float val = acc[i][j][v] + bc;
                if (ADD_RES) val += res[(long long)row * N + col];
                if (RELU) val = fmaxf(val, 0.f);
                if (OUT_F32) outF[(long long)row * N + col] = val;
                else         outB[(long long)row * N + col] = f2bf(val);
            }
        }
    }
}

// ---------------------------------------------------------------------------
// Flash attention: one wave handles a 16-row Q tile for one (b,h).
// q,k: bf16 [B,L,D] (head h at columns h*HD..); vT: bf16 [B,H,HD,L].
// scores = (Q K^T)*scale + rel_pos_bias, causal mask, online softmax, O = P V.
// ---------------------------------------------------------------------------
__global__ __launch_bounds__(32) void tb_attention(const bf16_t* __restrict__ q,
                                                   const bf16_t* __restrict__ k,
                                                   const bf16_t* __restrict__ vT,
                                                   const float* __restrict__ bias,
                                                   bf16_t* __restrict__ attn) {
    __shared__ bf16_t Pl[16][32];

    const int lane = threadIdx.x;
    const int half = lane >> 4;
    const int lr   = lane & 15;
    const int qt = blockIdx.x, h = blockIdx.y, bb = blockIdx.z;
    const int q0 = qt * 16;
    const float scale = 0.125f;  // 1/sqrt(64)

    v16bf qf[2];
#pragma unroll
    for (int j = 0; j < 2; ++j) {
        const bf16_t* qp = q + ((long long)bb * TB_L + q0 + lr) * TB_D + h * TB_HD + j * 32;
        union { v16bf v; v8bf hh[2]; } u;
        u.hh[0] = *(const v8bf*)(qp + half * 16);
        u.hh[1] = *(const v8bf*)(qp + half * 16 + 8);
        qf[j] = u.v;
    }

    v8f O[4];
#pragma unroll
    for (int n = 0; n < 4; ++n) O[n] = (v8f){0.f, 0.f, 0.f, 0.f, 0.f, 0.f, 0.f, 0.f};
    float m_[8], l_[8];
#pragma unroll
    for (int v = 0; v < 8; ++v) { m_[v] = -__builtin_inff(); l_[v] = 0.f; }

    const int kEnd = q0 + 16;
    for (int kc = 0; kc < kEnd; kc += 32) {
        v8f S0 = (v8f){0.f, 0.f, 0.f, 0.f, 0.f, 0.f, 0.f, 0.f};
        v8f S1 = (v8f){0.f, 0.f, 0.f, 0.f, 0.f, 0.f, 0.f, 0.f};
#pragma unroll
        for (int j = 0; j < 2; ++j) {
            v16bf kb0 = *(const v16bf*)(k + ((long long)bb * TB_L + kc + lr) * TB_D +
                                        h * TB_HD + j * 32 + half * 16);
            S0 = wmma_bf16(qf[j], kb0, S0);
            v16bf kb1 = *(const v16bf*)(k + ((long long)bb * TB_L + kc + 16 + lr) * TB_D +
                                        h * TB_HD + j * 32 + half * 16);
            S1 = wmma_bf16(qf[j], kb1, S1);
        }

        float sv0[8], sv1[8], cmax[8];
#pragma unroll
        for (int v = 0; v < 8; ++v) {
            int qrow = q0 + v + half * 8;
            const float* bp = bias + ((long long)h * TB_L + qrow) * TB_L;
            int key0 = kc + lr, key1 = kc + 16 + lr;
            float s0 = S0[v] * scale + bp[key0];
            float s1 = S1[v] * scale + bp[key1];
            if (key0 > qrow) s0 = -__builtin_inff();
            if (key1 > qrow) s1 = -__builtin_inff();
            sv0[v] = s0; sv1[v] = s1;
            cmax[v] = fmaxf(s0, s1);
        }
#pragma unroll
        for (int off = 1; off < 16; off <<= 1)
#pragma unroll
            for (int v = 0; v < 8; ++v)
                cmax[v] = fmaxf(cmax[v], __shfl_xor(cmax[v], off, 16));

        float psum[8];
#pragma unroll
        for (int v = 0; v < 8; ++v) {
            float mn = fmaxf(m_[v], cmax[v]);
            float alpha = __expf(m_[v] - mn);
            m_[v] = mn;
            float p0 = __expf(sv0[v] - mn);
            float p1 = __expf(sv1[v] - mn);
            psum[v] = p0 + p1;
            Pl[v + half * 8][lr]      = f2bf(p0);
            Pl[v + half * 8][16 + lr] = f2bf(p1);
            l_[v] *= alpha;
#pragma unroll
            for (int n = 0; n < 4; ++n) O[n][v] *= alpha;
        }
#pragma unroll
        for (int off = 1; off < 16; off <<= 1)
#pragma unroll
            for (int v = 0; v < 8; ++v)
                psum[v] += __shfl_xor(psum[v], off, 16);
#pragma unroll
        for (int v = 0; v < 8; ++v) l_[v] += psum[v];

        __syncthreads();
        union { v16bf v; v8bf hh[2]; } up;
        up.hh[0] = *(const v8bf*)(&Pl[lr][half * 16]);
        up.hh[1] = *(const v8bf*)(&Pl[lr][half * 16 + 8]);
        v16bf pf = up.v;
#pragma unroll
        for (int n = 0; n < 4; ++n) {
            v16bf vb = *(const v16bf*)(vT + (((long long)bb * TB_H + h) * TB_HD +
                                             n * 16 + lr) * TB_L + kc + half * 16);
            O[n] = wmma_bf16(pf, vb, O[n]);
        }
        __syncthreads();
    }

#pragma unroll
    for (int n = 0; n < 4; ++n) {
#pragma unroll
        for (int v = 0; v < 8; ++v) {
            int qrow = q0 + v + half * 8;
            int col = h * TB_HD + n * 16 + lr;
            attn[((long long)bb * TB_L + qrow) * TB_D + col] = f2bf(O[n][v] / l_[v]);
        }
    }
}

// ---------------------------------------------------------------------------
extern "C" void kernel_launch(void* const* d_in, const int* in_sizes, int n_in,
                              void* d_out, int out_size, void* d_ws, size_t ws_size,
                              hipStream_t stream) {
    const float* x    = (const float*)d_in[0];
    const float* relb = (const float*)d_in[1];
    const float* wq   = (const float*)d_in[2];
    const float* bq   = (const float*)d_in[3];
    const float* wk   = (const float*)d_in[4];
    const float* bk   = (const float*)d_in[5];
    const float* wv   = (const float*)d_in[6];
    const float* bv   = (const float*)d_in[7];
    const float* wo   = (const float*)d_in[8];
    const float* bo   = (const float*)d_in[9];
    const float* g1   = (const float*)d_in[10];
    const float* be1  = (const float*)d_in[11];
    const float* g2   = (const float*)d_in[12];
    const float* be2  = (const float*)d_in[13];
    const float* w1   = (const float*)d_in[14];
    const float* bf1  = (const float*)d_in[15];
    const float* w2   = (const float*)d_in[16];
    const float* bf2  = (const float*)d_in[17];

    size_t off = 0;
    auto wsAlloc = [&](size_t bytes) -> void* {
        void* p = (char*)d_ws + off;
        off += (bytes + 255) & ~(size_t)255;
        return p;
    };
    bf16_t* wqT  = (bf16_t*)wsAlloc((size_t)TB_D * TB_D * 2);
    bf16_t* wkT  = (bf16_t*)wsAlloc((size_t)TB_D * TB_D * 2);
    bf16_t* wvT  = (bf16_t*)wsAlloc((size_t)TB_D * TB_D * 2);
    bf16_t* woT  = (bf16_t*)wsAlloc((size_t)TB_D * TB_D * 2);
    bf16_t* w1T  = (bf16_t*)wsAlloc((size_t)TB_D * TB_FFN * 2);
    bf16_t* w2T  = (bf16_t*)wsAlloc((size_t)TB_FFN * TB_D * 2);
    bf16_t* xn   = (bf16_t*)wsAlloc((size_t)TB_M * TB_D * 2);
    bf16_t* attnb= (bf16_t*)wsAlloc((size_t)TB_M * TB_D * 2);
    float*  x2   = (float*) wsAlloc((size_t)TB_M * TB_D * 4);
    bf16_t* qb   = (bf16_t*)wsAlloc((size_t)TB_M * TB_D * 2);
    bf16_t* kb   = (bf16_t*)wsAlloc((size_t)TB_M * TB_D * 2);
    bf16_t* vb   = (bf16_t*)wsAlloc((size_t)TB_M * TB_D * 2);
    bf16_t* vTb  = (bf16_t*)wsAlloc((size_t)TB_M * TB_D * 2);
    bf16_t* hb   = qb;  // FFN hidden (64MB) reuses dead q/k/v/vT region

    // 1) weights -> bf16 transposed [N,K]
    {
        long long tot = (long long)TB_D * TB_D;
        int blocks = (int)((tot + 255) / 256);
        tb_convert_wT<<<blocks, 256, 0, stream>>>(wq, wqT, TB_D, TB_D);
        tb_convert_wT<<<blocks, 256, 0, stream>>>(wk, wkT, TB_D, TB_D);
        tb_convert_wT<<<blocks, 256, 0, stream>>>(wv, wvT, TB_D, TB_D);
        tb_convert_wT<<<blocks, 256, 0, stream>>>(wo, woT, TB_D, TB_D);
        long long totf = (long long)TB_D * TB_FFN;
        int blocksf = (int)((totf + 255) / 256);
        tb_convert_wT<<<blocksf, 256, 0, stream>>>(w1, w1T, TB_D, TB_FFN);
        tb_convert_wT<<<blocksf, 256, 0, stream>>>(w2, w2T, TB_FFN, TB_D);
    }

    // 2) LN1
    tb_layernorm<<<TB_M, 256, 0, stream>>>(x, g1, be1, xn);

    // 3) QKV projections (bf16 out)
    dim3 gD(TB_D / 128, TB_M / 128);
    tb_gemm<false, false, false><<<gD, 256, 0, stream>>>(xn, wqT, bq, nullptr, nullptr, qb, TB_M, TB_D, TB_D);
    tb_gemm<false, false, false><<<gD, 256, 0, stream>>>(xn, wkT, bk, nullptr, nullptr, kb, TB_M, TB_D, TB_D);
    tb_gemm<false, false, false><<<gD, 256, 0, stream>>>(xn, wvT, bv, nullptr, nullptr, vb, TB_M, TB_D, TB_D);

    // 4) V transpose to [B,H,HD,L]
    {
        long long tot = (long long)TB_M * TB_D;
        tb_transpose_v<<<(int)((tot + 255) / 256), 256, 0, stream>>>(vb, vTb);
    }

    // 5) attention
    tb_attention<<<dim3(TB_L / 16, TB_H, TB_B), 32, 0, stream>>>(qb, kb, vTb, relb, attnb);

    // 6) output projection + residual -> x2 (fp32)
    tb_gemm<true, false, true><<<gD, 256, 0, stream>>>(attnb, woT, bo, x, x2, nullptr, TB_M, TB_D, TB_D);

    // 7) LN2 (reuse xn buffer)
    tb_layernorm<<<TB_M, 256, 0, stream>>>(x2, g2, be2, xn);

    // 8) FFN up + ReLU (bf16 out)
    dim3 gF(TB_FFN / 128, TB_M / 128);
    tb_gemm<false, true, false><<<gF, 256, 0, stream>>>(xn, w1T, bf1, nullptr, nullptr, hb, TB_M, TB_FFN, TB_D);

    // 9) FFN down + residual -> out (fp32)
    tb_gemm<true, false, true><<<gD, 256, 0, stream>>>(hb, w2T, bf2, x2, (float*)d_out, nullptr, TB_M, TB_D, TB_FFN);
}
